// PAI_2104533975844
// MI455X (gfx1250) — matrix operand order repaired and checked by reading, not compiled, not run on hardware
//
#include <hip/hip_runtime.h>
#include <hip/hip_bf16.h>
#include <cstdint>
#include <cstddef>

#define BB  8
#define HH  240
#define WW  320
#define NFC 64

typedef __bf16       v16bf __attribute__((ext_vector_type(16)));
typedef float        v8f   __attribute__((ext_vector_type(8)));
typedef unsigned int v8u   __attribute__((ext_vector_type(8)));

// ---------- bf16 helpers (RNE) ----------
__device__ __forceinline__ uint32_t f2bf_bits(float f) {
  uint32_t u = __builtin_bit_cast(uint32_t, f);
  return (u + 0x7FFFu + ((u >> 16) & 1u)) >> 16;
}
__device__ __forceinline__ uint32_t pack_bf16(float a, float b) {
  return (f2bf_bits(a) & 0xFFFFu) | (f2bf_bits(b) << 16);
}
__device__ __forceinline__ float bf16lo(uint32_t u) { return __builtin_bit_cast(float, u << 16); }
__device__ __forceinline__ float bf16hi(uint32_t u) { return __builtin_bit_cast(float, u & 0xFFFF0000u); }

__device__ __forceinline__ float safe_recip(float d) {
  float mag = fmaxf(fabsf(d), 1e-4f);
  return (d >= 0.f ? 1.f : -1.f) / mag;
}
__device__ __forceinline__ float sigmoidf(float x) { return 1.f / (1.f + expf(-x)); }

// ---------- weight pre-pack: OIHW f32 -> [tap][cpair][o] packed bf16x2 ----------
__global__ void k_pack_w(const float* __restrict__ w, uint32_t* __restrict__ out, int cout) {
  int idx = blockIdx.x * 256 + threadIdx.x;
  int total = 9 * 32 * cout;
  if (idx >= total) return;
  int o   = idx % cout;
  int p   = (idx / cout) % 32;
  int tap = idx / (cout * 32);
  int ky = tap / 3, kx = tap % 3;
  const float* wb = w + ((size_t)o * NFC + 2 * p) * 9 + ky * 3 + kx;
  out[idx] = pack_bf16(wb[0], wb[9]);   // channel stride in OIHW = 9 floats
}

// ---------- implicit-GEMM WMMA conv, Cin=64, 3x3, pad 1 ----------
// Tile: 32x4 output pixels per 256-thread block (8 waves, wave32).
// Each wave owns ONE 16-out-channel N-tile and NT M-groups (16 pixels each);
// B fragments are built once per (tap,kblock) and reused across the wave's
// groups. Every wave always carries TWO independent WMMA accumulator chains
// (NT==2: one per M-group; NT==1: one per K-half, summed in the epilogue),
// so the XDL latency is hidden without hazard NOPs. Tap loop fully unrolled
// so each accumulator stays in a fixed register range (no v_mov copies).
#define EPI_BNELU     0
#define EPI_LEAKY     1
#define EPI_BIAS_NCHW 2

template<bool IN_BF16, int COUT, int EPI>
__global__ __launch_bounds__(256) void k_conv64(
    const void* __restrict__ in_, const uint32_t* __restrict__ wpack,
    const float* __restrict__ bias,
    const float* __restrict__ bng, const float* __restrict__ bnb,
    const float* __restrict__ bnm, const float* __restrict__ bnv,
    void* __restrict__ out_, int ctot, int cbase) {
  __shared__ uint32_t s_in[204 * 33];          // (4+2)x(32+2) pixels * 32 ch-pairs (+1 pad) = 26,928 B
  __shared__ uint32_t s_w[9 * 32 * COUT];      // COUT=32 -> 36,864 B ; total <= 63,792 B

  const int tid = threadIdx.x;
  const int blk = blockIdx.x;
  const int tx = blk % 10, ty = (blk / 10) % 60, b = blk / 600;
  const int x0t = tx * 32, y0t = ty * 4;

  for (int i = tid; i < 9 * 32 * COUT; i += 256) s_w[i] = wpack[i];

  const float*    inf = (const float*)in_;
  const uint32_t* inb = (const uint32_t*)in_;  // NHWC 64ch bf16 = 32 dwords/pixel
  for (int i = tid; i < 204 * 32; i += 256) {
    int pix = i >> 5, p = i & 31;
    int iy = pix / 34, ix = pix % 34;
    int gy = y0t - 1 + iy, gx = x0t - 1 + ix;
    uint32_t val = 0u;
    if (gy >= 0 && gy < HH && gx >= 0 && gx < WW) {
      if (IN_BF16) {
        val = inb[((size_t)(b * HH + gy) * WW + gx) * 32 + p];
      } else {
        size_t base = ((size_t)(b * NFC + 2 * p) * HH + gy) * WW + gx;
        val = pack_bf16(inf[base], inf[base + (size_t)HH * WW]);
      }
    }
    s_in[pix * 33 + p] = val;
  }
  __syncthreads();

  const int lane = tid & 31, wv = tid >> 5;
  const int mrow = lane & 15, hlf = lane >> 4;
  constexpr int NT  = COUT / 16;   // N-tiles total == M-groups per wave
  constexpr int GST = 8 / NT;      // group stride between a wave's M-groups
  const int nt = wv % NT;          // this wave's fixed out-channel tile
  const int gb = wv / NT;          // this wave's base M-group
  const int ocol = nt * 16 + mrow; // N column = out channel

  v8f acc[2];                      // always 2 independent WMMA chains
  acc[0] = (v8f){0.f, 0.f, 0.f, 0.f, 0.f, 0.f, 0.f, 0.f};
  acc[1] = (v8f){0.f, 0.f, 0.f, 0.f, 0.f, 0.f, 0.f, 0.f};

#pragma unroll
  for (int tap = 0; tap < 9; ++tap) {
    int ky = tap / 3, kx = tap - ky * 3;
#pragma unroll
    for (int kb = 0; kb < 2; ++kb) {
      int c2 = kb * 16;
      v8u bu;                                  // B: pair idx = i + 8*half
#pragma unroll
      for (int i = 0; i < 8; ++i)
        bu[i] = s_w[(tap * 32 + c2 + 8 * hlf + i) * COUT + ocol];
#pragma unroll
      for (int gi = 0; gi < NT; ++gi) {
        int g = gb + gi * GST;
        int py = g >> 1, px0 = (g & 1) * 16;
        int abase = ((py + ky) * 34 + px0 + mrow + kx) * 33;
        v8u au;                                // A: pair idx = i + 8*half
#pragma unroll
        for (int i = 0; i < 8; ++i)
          au[i] = s_in[abase + c2 + 8 * hlf + i];
        int ai = (NT == 1) ? kb : gi;          // chain select: K-half or M-group
        acc[ai] = __builtin_amdgcn_wmma_f32_16x16x32_bf16(
            false, __builtin_bit_cast(v16bf, au),
            false, __builtin_bit_cast(v16bf, bu),
            (short)0, acc[ai], false, false);
      }
    }
  }

  // epilogue (per-channel affine computed once; C/D: M = r + 8*half, N = lane&15)
  float scale = 1.f, shift = bias[ocol];
  if constexpr (EPI == EPI_BNELU) {
    float inv = bng[ocol] / sqrtf(bnv[ocol] + 1e-5f);
    shift = (bias[ocol] - bnm[ocol]) * inv + bnb[ocol];
    scale = inv;
  }
#pragma unroll
  for (int gi = 0; gi < NT; ++gi) {
    int g = gb + gi * GST;
    int py = g >> 1, px0 = (g & 1) * 16;
    int yy = y0t + py;
#pragma unroll
    for (int r = 0; r < 8; ++r) {
      float a = (NT == 1) ? (acc[0][r] + acc[1][r]) : acc[gi][r];
      float vv = a * scale + shift;
      if constexpr (EPI == EPI_BNELU) vv = vv > 0.f ? vv : (expf(vv) - 1.f);
      if constexpr (EPI == EPI_LEAKY) vv = vv >= 0.f ? vv : 0.1f * vv;
      int xx = x0t + px0 + r + 8 * hlf;
      if constexpr (EPI == EPI_BIAS_NCHW) {
        ((float*)out_)[((size_t)(b * COUT + ocol) * HH + yy) * WW + xx] = vv;
      } else {
        ((uint16_t*)out_)[((size_t)(b * HH + yy) * WW + xx) * ctot + cbase + ocol] =
            (uint16_t)f2bf_bits(vv);
      }
    }
  }
}

// ---------- small conv: Cin=16 NHWC-bf16 -> 3 or 4 ch, fused epilogues ----------
template<int COUT, bool IS_PD>
__global__ __launch_bounds__(256) void k_conv16(
    const uint16_t* __restrict__ tin, const float* __restrict__ w,
    const float* __restrict__ bias, float* __restrict__ outA, float* __restrict__ outB) {
  __shared__ float sw[COUT * 16 * 9];
  int tid = threadIdx.x;
  for (int i = tid; i < COUT * 16 * 9; i += 256) sw[i] = w[i];   // [co][c][tap] == OIHW flat
  __syncthreads();
  int idx = blockIdx.x * 256 + tid;
  if (idx >= BB * HH * WW) return;
  int x = idx % WW, y = (idx / WW) % HH, b = idx / (WW * HH);

  float acc[COUT];
#pragma unroll
  for (int co = 0; co < COUT; ++co) acc[co] = bias[co];

  for (int ky = 0; ky < 3; ++ky) {
    int gy = y + ky - 1; if (gy < 0 || gy >= HH) continue;
    for (int kx = 0; kx < 3; ++kx) {
      int gx = x + kx - 1; if (gx < 0 || gx >= WW) continue;
      int tap = ky * 3 + kx;
      const uint32_t* pp = (const uint32_t*)(tin + ((size_t)(b * HH + gy) * WW + gx) * 16);
      float v[16];
#pragma unroll
      for (int j = 0; j < 8; ++j) { uint32_t u = pp[j]; v[2 * j] = bf16lo(u); v[2 * j + 1] = bf16hi(u); }
#pragma unroll
      for (int co = 0; co < COUT; ++co) {
        float a = acc[co];
#pragma unroll
        for (int c = 0; c < 16; ++c) a += v[c] * sw[(co * 16 + c) * 9 + tap];
        acc[co] = a;
      }
    }
  }
  if constexpr (!IS_PD) {     // offset head: conf + duv
    outA[idx] = sigmoidf(acc[0]);
    float* dd = outB + (size_t)b * 2 * HH * WW;
    dd[y * WW + x]            = tanhf(acc[1]) * 0.1f;
    dd[HH * WW + y * WW + x]  = tanhf(acc[2]) * 0.1f;
  } else {                    // plane head: normalized fgh + sigmoid s (NHWC float4)
    float n2 = acc[0] * acc[0] + acc[1] * acc[1] + acc[2] * acc[2];
    float inv = 1.f / sqrtf(n2 + 1e-12f);
    float4 r; r.x = acc[0] * inv; r.y = acc[1] * inv; r.z = acc[2] * inv; r.w = sigmoidf(acc[3]);
    ((float4*)outA)[idx] = r;
  }
}

// ---------- one duv refinement step: duv += grid_sample(duv, clip(coords+duv), zeros) ----------
__global__ __launch_bounds__(256) void k_refine(const float* __restrict__ din,
                                                float* __restrict__ dout) {
  int idx = blockIdx.x * 256 + threadIdx.x;
  if (idx >= BB * HH * WW) return;
  int x = idx % WW, y = (idx / WW) % HH, b = idx / (WW * HH);
  const float* d0 = din + (size_t)b * 2 * HH * WW;
  float du = d0[y * WW + x], dv = d0[HH * WW + y * WW + x];
  float U = -1.f + 2.f * x / (WW - 1);
  float V = -1.f + 2.f * y / (HH - 1);
  float ox = fminf(fmaxf(U + du, -1.f), 1.f);
  float oy = fminf(fmaxf(V + dv, -1.f), 1.f);
  float gx = (ox + 1.f) * 0.5f * (WW - 1);
  float gy = (oy + 1.f) * 0.5f * (HH - 1);
  float xf = floorf(gx), yf = floorf(gy);
  float wx = gx - xf, wy = gy - yf;
  float s0 = 0.f, s1 = 0.f;
#pragma unroll
  for (int cy = 0; cy < 2; ++cy)
#pragma unroll
    for (int cx = 0; cx < 2; ++cx) {
      float xi = xf + cx, yi = yf + cy;
      float wgt = (cx ? wx : 1.f - wx) * (cy ? wy : 1.f - wy);
      bool ok = (xi >= 0.f) && (xi <= (float)(WW - 1)) && (yi >= 0.f) && (yi <= (float)(HH - 1));
      int xc = (int)fminf(fmaxf(xi, 0.f), (float)(WW - 1));
      int yc = (int)fminf(fmaxf(yi, 0.f), (float)(HH - 1));
      float f = ok ? wgt : 0.f;
      s0 += f * d0[yc * WW + xc];
      s1 += f * d0[HH * WW + yc * WW + xc];
    }
  float* o0 = dout + (size_t)b * 2 * HH * WW;
  o0[y * WW + x]           = du + s0;
  o0[HH * WW + y * WW + x] = dv + s1;
}

// ---------- plane disparity + border grid-sample + confidence blend ----------
__global__ __launch_bounds__(256) void k_combine(
    const float* __restrict__ conf, const float4* __restrict__ fghs,
    const float* __restrict__ duv, float* __restrict__ dfr) {
  int idx = blockIdx.x * 256 + threadIdx.x;
  if (idx >= BB * HH * WW) return;
  int x = idx % WW, y = (idx / WW) % HH, b = idx / (WW * HH);
  float U = -1.f + 2.f * x / (WW - 1);
  float V = -1.f + 2.f * y / (HH - 1);
  float4 fs = fghs[idx];
  float disp = (fs.x * U + fs.y * V + fs.z) / (fs.w + 1e-8f);
  float r2d = safe_recip(disp);

  const float* d0 = duv + (size_t)b * 2 * HH * WW;
  float du = d0[y * WW + x], dv = d0[HH * WW + y * WW + x];
  float ox = fminf(fmaxf(U + du, -1.f), 1.f);
  float oy = fminf(fmaxf(V + dv, -1.f), 1.f);
  float gx = (ox + 1.f) * 0.5f * (WW - 1);
  float gy = (oy + 1.f) * 0.5f * (HH - 1);
  float xf = floorf(gx), yf = floorf(gy);
  float wx = gx - xf, wy = gy - yf;
  const float4* fb = fghs + (size_t)b * HH * WW;
  float ax = 0.f, ay = 0.f, az = 0.f, aw = 0.f;
#pragma unroll
  for (int cy = 0; cy < 2; ++cy)
#pragma unroll
    for (int cx = 0; cx < 2; ++cx) {
      float wgt = (cx ? wx : 1.f - wx) * (cy ? wy : 1.f - wy);
      int xc = (int)fminf(fmaxf(xf + cx, 0.f), (float)(WW - 1));
      int yc = (int)fminf(fmaxf(yf + cy, 0.f), (float)(HH - 1));
      float4 s = fb[yc * WW + xc];                 // border padding: clamp, no mask
      ax += wgt * s.x; ay += wgt * s.y; az += wgt * s.z; aw += wgt * s.w;
    }
  float disp_o = (ax * U + ay * V + az) / (aw + 1e-8f);
  float doff = safe_recip(disp_o);
  float c = conf[idx];
  dfr[idx] = (1.f - c) * r2d + c * doff;
}

// ---------- 1->32 conv + leaky, writes channels 32..63 of NHWC-bf16 feat ----------
__global__ __launch_bounds__(256) void k_featrgb(
    const float* __restrict__ dfr, const float* __restrict__ w,
    const float* __restrict__ bias, uint16_t* __restrict__ feat) {
  __shared__ float sw[32 * 9];
  __shared__ float sb[32];
  int tid = threadIdx.x;
  if (tid < 32 * 9) sw[tid] = w[tid];
  if (tid < 32) sb[tid] = bias[tid];
  __syncthreads();
  int idx = blockIdx.x * 256 + tid;
  if (idx >= BB * HH * WW) return;
  int x = idx % WW, y = (idx / WW) % HH, b = idx / (WW * HH);
  const float* src = dfr + (size_t)b * HH * WW;
  float v[9];
#pragma unroll
  for (int ky = 0; ky < 3; ++ky)
#pragma unroll
    for (int kx = 0; kx < 3; ++kx) {
      int gy = y + ky - 1, gx = x + kx - 1;
      bool in = (gy >= 0 && gy < HH && gx >= 0 && gx < WW);
      v[ky * 3 + kx] = in ? src[gy * WW + gx] : 0.f;
    }
  uint32_t* op = (uint32_t*)(feat + ((size_t)(b * HH + y) * WW + x) * 64) + 16;
#pragma unroll
  for (int k = 0; k < 16; ++k) {
    float a0 = sb[2 * k], a1 = sb[2 * k + 1];
#pragma unroll
    for (int tap = 0; tap < 9; ++tap) {
      a0 += v[tap] * sw[(2 * k) * 9 + tap];
      a1 += v[tap] * sw[(2 * k + 1) * 9 + tap];
    }
    a0 = a0 >= 0.f ? a0 : 0.1f * a0;
    a1 = a1 >= 0.f ? a1 : 0.1f * a1;
    op[k] = pack_bf16(a0, a1);
  }
}

// ============================================================================
extern "C" void kernel_launch(void* const* d_in, const int* in_sizes, int n_in,
                              void* d_out, int out_size, void* d_ws, size_t ws_size,
                              hipStream_t stream) {
  (void)in_sizes; (void)n_in; (void)out_size; (void)ws_size;
  const float* depth  = (const float*)d_in[0];
  const float* rgb    = (const float*)d_in[1];
  const float* w_off1 = (const float*)d_in[2];
  const float* b_off1 = (const float*)d_in[3];
  const float* g_off  = (const float*)d_in[4];
  const float* be_off = (const float*)d_in[5];
  const float* m_off  = (const float*)d_in[6];
  const float* v_off  = (const float*)d_in[7];
  const float* w_off2 = (const float*)d_in[8];
  const float* b_off2 = (const float*)d_in[9];
  const float* w_pd1  = (const float*)d_in[10];
  const float* b_pd1  = (const float*)d_in[11];
  const float* g_pd   = (const float*)d_in[12];
  const float* be_pd  = (const float*)d_in[13];
  const float* m_pd   = (const float*)d_in[14];
  const float* v_pd   = (const float*)d_in[15];
  const float* w_pd2  = (const float*)d_in[16];
  const float* b_pd2  = (const float*)d_in[17];
  const float* w_d2f  = (const float*)d_in[18];
  const float* b_d2f  = (const float*)d_in[19];
  const float* w_dm   = (const float*)d_in[20];
  const float* b_dm   = (const float*)d_in[21];
  const float* w_fus  = (const float*)d_in[22];
  const float* b_fus  = (const float*)d_in[23];

  char* ws = (char*)d_ws;
  size_t off = 0;
  auto take = [&](size_t bytes) -> char* {
    char* p = ws + off;
    off = (off + bytes + 255) & ~(size_t)255;
    return p;
  };
  const size_t NPIX = (size_t)BB * HH * WW;      // 614,400
  uint32_t* wp_off1 = (uint32_t*)take(9 * 32 * 16 * 4);
  uint32_t* wp_pd1  = (uint32_t*)take(9 * 32 * 16 * 4);
  uint32_t* wp_dm   = (uint32_t*)take(9 * 32 * 32 * 4);
  uint32_t* wp_fus  = (uint32_t*)take(9 * 32 * 32 * 4);
  uint16_t* t_off   = (uint16_t*)take(NPIX * 16 * 2);   // NHWC bf16
  uint16_t* t_pd    = (uint16_t*)take(NPIX * 16 * 2);   // NHWC bf16
  float*    conf    = (float*)take(NPIX * 4);
  float*    duv_a   = (float*)take(NPIX * 2 * 4);       // NCHW 2ch
  float*    duv_b   = (float*)take(NPIX * 2 * 4);
  float*    fghs    = (float*)take(NPIX * 4 * 4);       // NHWC float4
  float*    dfr     = (float*)take(NPIX * 4);
  uint16_t* feat    = (uint16_t*)take(NPIX * 64 * 2);   // NHWC bf16, 64ch concat

  const int PXB = (int)((NPIX + 255) / 256);            // 2400
  const int CVB = BB * 60 * 10;                         // 4800 (32x4 tiles)

  // weight packing (tiny)
  k_pack_w<<<18, 256, 0, stream>>>(w_off1, wp_off1, 16);
  k_pack_w<<<18, 256, 0, stream>>>(w_pd1,  wp_pd1,  16);
  k_pack_w<<<36, 256, 0, stream>>>(w_dm,   wp_dm,   32);
  k_pack_w<<<36, 256, 0, stream>>>(w_fus,  wp_fus,  32);

  // offset branch: conv64->BN+ELU (WMMA), then 16->3 head -> conf + duv
  k_conv64<false, 16, EPI_BNELU><<<CVB, 256, 0, stream>>>(
      depth, wp_off1, b_off1, g_off, be_off, m_off, v_off, t_off, 16, 0);
  k_conv16<3, false><<<PXB, 256, 0, stream>>>(t_off, w_off2, b_off2, conf, duv_a);

  // plane branch: conv64->BN+ELU (WMMA), then 16->4 head -> fghs
  k_conv64<false, 16, EPI_BNELU><<<CVB, 256, 0, stream>>>(
      rgb, wp_pd1, b_pd1, g_pd, be_pd, m_pd, v_pd, t_pd, 16, 0);
  k_conv16<4, true><<<PXB, 256, 0, stream>>>(t_pd, w_pd2, b_pd2, fghs, nullptr);

  // 3 sequential duv refinement steps (ping-pong)
  k_refine<<<PXB, 256, 0, stream>>>(duv_a, duv_b);
  k_refine<<<PXB, 256, 0, stream>>>(duv_b, duv_a);
  k_refine<<<PXB, 256, 0, stream>>>(duv_a, duv_b);

  // depth_FromRGB
  k_combine<<<PXB, 256, 0, stream>>>(conf, (const float4*)fghs, duv_b, dfr);

  // feat_d = leaky(conv64->32) into feat[0:32] (WMMA)
  k_conv64<false, 32, EPI_LEAKY><<<CVB, 256, 0, stream>>>(
      depth, wp_dm, b_dm, nullptr, nullptr, nullptr, nullptr, feat, 64, 0);
  // feat_rgb = leaky(conv 1->32) into feat[32:64]
  k_featrgb<<<PXB, 256, 0, stream>>>(dfr, w_d2f, b_d2f, feat);

  // fusion: conv 64->32 over NHWC-bf16 feat (WMMA) -> NCHW f32 output
  k_conv64<true, 32, EPI_BIAS_NCHW><<<CVB, 256, 0, stream>>>(
      feat, wp_fus, b_fus, nullptr, nullptr, nullptr, nullptr, d_out, 0, 0);
}
